// SelfAttention_55327768708644
// MI455X (gfx1250) — matrix operand order repaired and checked by reading, not compiled, not run on hardware
//
#include <hip/hip_runtime.h>
#include <hip/hip_bf16.h>

typedef __attribute__((ext_vector_type(16))) __bf16 v16bf;
typedef __attribute__((ext_vector_type(8)))  float  v8f;
typedef __attribute__((ext_vector_type(4)))  float  v4f;

#define Bb 2
#define Tt 512
#define Dd 768
#define Uu 768
#define IB 4   // query rows per block (reuse factor for ux / x streams)

// ---------------- conversion kernels ----------------
__global__ __launch_bounds__(256) void cvt_bf16(const float* __restrict__ in,
                                                __bf16* __restrict__ out, int n) {
  int i = blockIdx.x * 256 + threadIdx.x;
  if (i < n) out[i] = (__bf16)in[i];
}

// in: rows x cols (row-major), out: cols x rows (row-major) i.e. transposed
__global__ __launch_bounds__(256) void cvt_bf16_T(const float* __restrict__ in,
                                                  __bf16* __restrict__ out,
                                                  int rows, int cols) {
  int i = blockIdx.x * 256 + threadIdx.x;
  if (i < rows * cols) {
    int r = i / cols, c = i % cols;
    out[(size_t)c * rows + r] = (__bf16)in[i];
  }
}

// ---------------- fused dual-projection WMMA GEMM ----------------
// Yw[M,N] = X[M,K] * Wt[N,K]^T ; Yu[M,N] = X[M,K] * Ut[N,K]^T + bias
// One wave per 16x16 tile pair; A fragment reused for both B matrices.
union Frag16 { v16bf v; unsigned u[8]; };

__global__ __launch_bounds__(256)
void proj_dual_wmma(const __bf16* __restrict__ X, const __bf16* __restrict__ Wt,
                    const __bf16* __restrict__ Ut, const float* __restrict__ bias,
                    float* __restrict__ Yw, float* __restrict__ Yu,
                    int M, int N, int K) {
  int lane = threadIdx.x & 31;
  int wave = threadIdx.x >> 5;
  int tilesN = N >> 4;
  int tile = blockIdx.x * 8 + wave;
  int tm = tile / tilesN;
  int tn = tile % tilesN;
  if (tm >= (M >> 4)) return;            // wave-uniform guard (EXEC stays all-1s)

  int hi = lane >> 4;
  int lr = lane & 15;
  const __bf16* arow = X  + (size_t)(tm * 16 + lr) * K;
  const __bf16* wrow = Wt + (size_t)(tn * 16 + lr) * K;
  const __bf16* urow = Ut + (size_t)(tn * 16 + lr) * K;

  v8f accw = {}, accu = {};
  for (int kk = 0; kk < K; kk += 32) {
    Frag16 a, bw, bu;
    int ka = kk + hi * 16;
#pragma unroll
    for (int v = 0; v < 4; ++v) {
      a.u[v]     = *(const unsigned*)(arow + ka + 2 * v);
      a.u[v + 4] = *(const unsigned*)(arow + ka + 8 + 2 * v);
    }
    int kb = kk + hi * 16;
#pragma unroll
    for (int v = 0; v < 8; ++v) bw.u[v] = *(const unsigned*)(wrow + kb + 2 * v);
#pragma unroll
    for (int v = 0; v < 8; ++v) bu.u[v] = *(const unsigned*)(urow + kb + 2 * v);
    accw = __builtin_amdgcn_wmma_f32_16x16x32_bf16(
        false, a.v, false, bw.v, (short)0, accw, false, false);
    accu = __builtin_amdgcn_wmma_f32_16x16x32_bf16(
        false, a.v, false, bu.v, (short)0, accu, false, false);
  }

  float bb = bias[tn * 16 + lr];
  size_t obase = (size_t)(tm * 16 + hi * 8) * N + tn * 16 + lr;
#pragma unroll
  for (int v = 0; v < 8; ++v) {
    Yw[obase + (size_t)v * N] = accw[v];
    Yu[obase + (size_t)v * N] = accu[v] + bb;
  }
}

// ---------------- fused score/tanh + softmax + context ----------------
__device__ __forceinline__ float fast_tanh(float x) {
#if __has_builtin(__builtin_amdgcn_tanhf)
  return __builtin_amdgcn_tanhf(x);            // v_tanh_f32 (CDNA5 trans op)
#elif __has_builtin(__builtin_amdgcn_tanh_f32)
  return __builtin_amdgcn_tanh_f32(x);
#else
  // tanh(x) = 1 - 2/(exp2(2*log2e*x)+1); saturates to +/-1 for large |x|
  float ex = __builtin_amdgcn_exp2f(x * 2.8853900817779268f);
  return 1.0f - 2.0f * __builtin_amdgcn_rcpf(ex + 1.0f);
#endif
}

__device__ __forceinline__ float wave_sum(float v) {
#pragma unroll
  for (int off = 16; off > 0; off >>= 1) v += __shfl_xor(v, off);
  return v;
}

// block-wide reductions over 16 waves (512 threads), s_r has 16 entries
__device__ __forceinline__ float block_max(float v, float* s_r, int tid) {
  int lane = tid & 31, wv = tid >> 5;
#pragma unroll
  for (int off = 16; off > 0; off >>= 1) v = fmaxf(v, __shfl_xor(v, off));
  __syncthreads();                 // protect s_r reuse across calls
  if (lane == 0) s_r[wv] = v;
  __syncthreads();
  float m = s_r[0];
#pragma unroll
  for (int k = 1; k < 16; ++k) m = fmaxf(m, s_r[k]);
  return m;
}

__device__ __forceinline__ float block_sum(float v, float* s_r, int tid) {
  int lane = tid & 31, wv = tid >> 5;
  v = wave_sum(v);
  __syncthreads();
  if (lane == 0) s_r[wv] = v;
  __syncthreads();
  float m = s_r[0];
#pragma unroll
  for (int k = 1; k < 16; ++k) m += s_r[k];
  return m;
}

__global__ __launch_bounds__(512)
void attn_kernel(const float* __restrict__ x, const float* __restrict__ wx,
                 const float* __restrict__ ux, const float* __restrict__ V_a,
                 float* __restrict__ out) {
  __shared__ alignas(16) float s_wx[IB][Uu];
  __shared__ alignas(16) float s_V[Uu];
  __shared__ alignas(16) float s_p[IB][Tt];
  __shared__ alignas(16) float s_r[16];

  int tid = threadIdx.x;
  int b  = blockIdx.x / (Tt / IB);
  int i0 = (blockIdx.x % (Tt / IB)) * IB;

  for (int r = 0; r < IB; ++r) {
    const float* wxrow = wx + ((size_t)b * Tt + i0 + r) * Uu;
    for (int u = tid; u < Uu; u += 512) s_wx[r][u] = wxrow[u];
  }
  for (int u = tid; u < Uu; u += 512) s_V[u] = V_a[u];
  __syncthreads();

  // scores: one wave per key j; each ux row feeds IB query rows (4x reuse)
  int wave = tid >> 5, lane = tid & 31;
  const float* uxb = ux + (size_t)b * Tt * Uu;
  for (int j = wave; j < Tt; j += 16) {
    const float* uxj = uxb + (size_t)j * Uu;
    float a0 = 0.0f, a1 = 0.0f, a2 = 0.0f, a3 = 0.0f;
#pragma unroll
    for (int u = lane * 4; u < Uu; u += 128) {
      v4f uv = *(const v4f*)(uxj + u);
      v4f Vv = *(const v4f*)(s_V + u);
      v4f w0 = *(const v4f*)(&s_wx[0][u]);
      v4f w1 = *(const v4f*)(&s_wx[1][u]);
      v4f w2 = *(const v4f*)(&s_wx[2][u]);
      v4f w3 = *(const v4f*)(&s_wx[3][u]);
#pragma unroll
      for (int c = 0; c < 4; ++c) {
        float uc = uv[c], vc = Vv[c];
        a0 = fmaf(vc, fast_tanh(w0[c] + uc), a0);
        a1 = fmaf(vc, fast_tanh(w1[c] + uc), a1);
        a2 = fmaf(vc, fast_tanh(w2[c] + uc), a2);
        a3 = fmaf(vc, fast_tanh(w3[c] + uc), a3);
      }
    }
    a0 = wave_sum(a0); a1 = wave_sum(a1); a2 = wave_sum(a2); a3 = wave_sum(a3);
    if (lane == 0) {
      s_p[0][j] = a0; s_p[1][j] = a1; s_p[2][j] = a2; s_p[3][j] = a3;
    }
  }
  __syncthreads();

  // softmax per query row; each thread owns exactly one key (Tt == 512)
#pragma unroll
  for (int r = 0; r < IB; ++r) {
    float e  = s_p[r][tid];
    float mx = block_max(e, s_r, tid);
    float p  = __builtin_amdgcn_exp2f((e - mx) * 1.4426950408889634f);
    float sm = block_sum(p, s_r, tid);
    s_p[r][tid] = p * __builtin_amdgcn_rcpf(sm);
  }
  __syncthreads();

  // context: each x[b,j,d] load feeds IB accumulators
  const float* xb = x + (size_t)b * Tt * Dd;
  float* obase = out + ((size_t)b * Tt + i0) * Dd;
  for (int d = tid; d < Dd; d += 512) {
    float c0 = 0.0f, c1 = 0.0f, c2 = 0.0f, c3 = 0.0f;
    for (int j = 0; j < Tt; ++j) {
      float xv = xb[(size_t)j * Dd + d];
      c0 = fmaf(s_p[0][j], xv, c0);
      c1 = fmaf(s_p[1][j], xv, c1);
      c2 = fmaf(s_p[2][j], xv, c2);
      c3 = fmaf(s_p[3][j], xv, c3);
    }
    obase[d]           = c0;
    obase[Dd + d]      = c1;
    obase[2 * Dd + d]  = c2;
    obase[3 * Dd + d]  = c3;
  }
}

// ---------------- launch ----------------
extern "C" void kernel_launch(void* const* d_in, const int* in_sizes, int n_in,
                              void* d_out, int out_size, void* d_ws, size_t ws_size,
                              hipStream_t stream) {
  const float* x   = (const float*)d_in[0];
  const float* V_a = (const float*)d_in[1];
  const float* U_a = (const float*)d_in[2];
  const float* b_u = (const float*)d_in[3];
  const float* W_a = (const float*)d_in[4];
  float* out = (float*)d_out;

  const int M = Bb * Tt;  // 1024 query rows total
  char* ws = (char*)d_ws;
  size_t off = 0;
  __bf16* xb = (__bf16*)(ws + off); off += (size_t)M * Dd * sizeof(__bf16);
  __bf16* Wt = (__bf16*)(ws + off); off += (size_t)Dd * Uu * sizeof(__bf16);
  __bf16* Ut = (__bf16*)(ws + off); off += (size_t)Dd * Uu * sizeof(__bf16);
  float*  wx = (float*) (ws + off); off += (size_t)M * Uu * sizeof(float);
  float*  ux = (float*) (ws + off); off += (size_t)M * Uu * sizeof(float);

  int nx = M * Dd;
  cvt_bf16  <<<(nx + 255) / 256, 256, 0, stream>>>(x, xb, nx);
  int nw = Dd * Uu;
  cvt_bf16_T<<<(nw + 255) / 256, 256, 0, stream>>>(W_a, Wt, Dd, Uu);
  cvt_bf16_T<<<(nw + 255) / 256, 256, 0, stream>>>(U_a, Ut, Dd, Uu);

  int tiles = (M / 16) * (Uu / 16);        // 3072 tiles, 8 waves/block
  proj_dual_wmma<<<tiles / 8, 256, 0, stream>>>(xb, Wt, Ut, b_u, wx, ux, M, Uu, Dd);

  attn_kernel<<<(Bb * Tt) / IB, 512, 0, stream>>>(x, wx, ux, V_a, out);
}
